// LiquidEnsembleLayer_36919538876929
// MI455X (gfx1250) — matrix operand order repaired and checked by reading, not compiled, not run on hardware
//
#include <hip/hip_runtime.h>
#include <hip/hip_bf16.h>

#define EPSV 0.01f

typedef __attribute__((ext_vector_type(16))) __bf16 v16bf;
typedef __attribute__((ext_vector_type(8)))  float  v8f;
typedef __attribute__((ext_vector_type(4)))  unsigned int v4u;

union FragBF { v4u q[2]; v16bf v; };
union PackB64 { __bf16 h[4]; unsigned long long u; };
union PackB32 { __bf16 h[2]; unsigned u; };

// ---------------------------------------------------------------------------
// Kernel 1: logits -> softmax -> reduced 16x16 solve -> power[B,16]  (all f32)
// ---------------------------------------------------------------------------
__global__ __launch_bounds__(256) void power_kernel(
    const float* __restrict__ x,   // [4096,1024]
    const float* __restrict__ Wd,  // [16,1024,16]
    const float* __restrict__ bd,  // [16,16]
    float* __restrict__ power)     // [4096,16]
{
    __shared__ float xs[4][1024];    // 16 KB: 4 x-rows (Wd reuse x4)
    __shared__ float lg[4][16][17];  // logits for 4 batches
    __shared__ float ge[32][289];    // 16x17 augmented system + 16 diag; stride 289 -> conflict-free

    const int tid = threadIdx.x;
    const int i = tid >> 4;          // citizen row of logit
    const int j = tid & 15;          // citizen col of logit
    const int b0 = blockIdx.x * 32;
    const float bdv = bd[i * 16 + j];

    for (int g = 0; g < 8; ++g) {
        __syncthreads();
        for (int idx = tid; idx < 4096; idx += 256) {
            int q = idx >> 10;
            xs[q][idx & 1023] = x[(size_t)(b0 + g * 4 + q) * 1024 + (idx & 1023)];
        }
        __syncthreads();
        float a0 = bdv, a1 = bdv, a2 = bdv, a3 = bdv;
        for (int d = 0; d < 1024; ++d) {
            float w = Wd[(size_t)i * 16384 + d * 16 + j];
            a0 = fmaf(xs[0][d], w, a0);
            a1 = fmaf(xs[1][d], w, a1);
            a2 = fmaf(xs[2][d], w, a2);
            a3 = fmaf(xs[3][d], w, a3);
        }
        lg[0][i][j] = a0; lg[1][i][j] = a1; lg[2][i][j] = a2; lg[3][i][j] = a3;
        __syncthreads();
        for (int q = 0; q < 4; ++q) {
            int bl = g * 4 + q;
            float mx = lg[q][i][0];
            for (int c = 1; c < 16; ++c) mx = fmaxf(mx, lg[q][i][c]);
            float sum = 0.0f;
            for (int c = 0; c < 16; ++c) sum += __expf(lg[q][i][c] - mx);
            float dij = __expf(lg[q][i][j] - mx) / sum;      // dall[b,i,j]
            ge[bl][j * 17 + i] = (i == j) ? 1.0f : -(1.0f - EPSV) * dij;
            if (i == j) ge[bl][272 + i] = dij;               // self-delegation diag
            if (j == 0) ge[bl][i * 17 + 16] = 1.0f;          // rhs = ones
        }
    }
    __syncthreads();
    if (tid < 32) {
        float* M = ge[tid];
        for (int k = 0; k < 16; ++k) {
            float piv = 1.0f / M[k * 17 + k];
            for (int r = k + 1; r < 16; ++r) {
                float f = M[r * 17 + k] * piv;
                for (int c = k + 1; c < 17; ++c)
                    M[r * 17 + c] = fmaf(-f, M[k * 17 + c], M[r * 17 + c]);
            }
        }
        float v[16];
        for (int r = 15; r >= 0; --r) {
            float s = M[r * 17 + 16];
            for (int c = r + 1; c < 16; ++c) s = fmaf(-M[r * 17 + c], v[c], s);
            v[r] = s / M[r * 17 + r];
        }
        float vsum = 0.0f;
        for (int c = 0; c < 16; ++c) vsum += v[c];
        float pen = 1.0f + EPSV * vsum;                      // sink power_ext
        const int b = b0 + tid;
        for (int c = 0; c < 16; ++c)
            power[b * 16 + c] = v[c] * (1.0f - EPSV) * M[272 + c]
                              + pen * (1.0f / 16.0f) - (1.0f / 16.0f);
    }
}

// ---------------------------------------------------------------------------
// Kernel 2: out[b,o] = sum_k (power[b,k>>10]*x[b,k&1023]) * Wy_flat[k,o] + bias
// bf16 WMMA 16x16x32 / f32 accum, 128x128 tile per block, 8 waves (2x4),
// each wave 4x2 fragments. Register-double-buffered global prefetch so HBM/L2
// latency overlaps the 16 WMMAs per K-tile.
// ---------------------------------------------------------------------------
__global__ __launch_bounds__(256) void gemm_kernel(
    const float* __restrict__ x,      // [4096,1024]
    const float* __restrict__ Wy,     // [16,1024,1024] == Wy_flat[16384,1024]
    const float* __restrict__ by,     // [16,1024]
    const float* __restrict__ power,  // [4096,16]
    float* __restrict__ out)          // [4096,1024]
{
    __shared__ __bf16 As[128 * 72];   // A tile, (m, k), padded stride 72
    __shared__ __bf16 Bs[128 * 72];   // B tile, transposed (o, k), padded stride 72

    const int tid  = threadIdx.x;
    const int wave = tid >> 5;
    const int lane = tid & 31;
    const int lm = lane & 15;
    const int lh = lane >> 4;
    const int m0 = blockIdx.x * 128;  // batch tile
    const int o0 = blockIdx.y * 128;  // output-feature tile
    const int wm = (wave >> 2) * 64;  // wave M offset (0/64)
    const int wn = (wave & 3) * 32;   // wave N offset (0/32/64/96)

    v8f acc[4][2] = {};

    // ---- register prefetch state ----
    float4 pa[8];  float pw[8];       // A: 128x64 f32 -> 8 float4/thread (+power)
    float4 pbl[4], pbh[4];            // B: 64x128 f32 -> 4 k-row-pairs/thread

    auto prefetch = [&](int k0) {
        const int xoff = k0 & 1023;
        const int nidx = k0 >> 10;
#pragma unroll
        for (int it = 0; it < 8; ++it) {
            int id  = tid + it * 256;
            int row = id >> 4;
            int c4  = (id & 15) << 2;
            int b   = m0 + row;
            pw[it] = power[b * 16 + nidx];
            pa[it] = *(const float4*)(x + (size_t)b * 1024 + xoff + c4);
        }
#pragma unroll
        for (int it = 0; it < 4; ++it) {
            int id = tid + it * 256;
            int kp = id >> 5;                 // k-pair index 0..31
            int c4 = (id & 31) << 2;          // o offset 0..124
            const float* base = Wy + (size_t)(k0 + 2 * kp) * 1024 + o0 + c4;
            pbl[it] = *(const float4*)(base);
            pbh[it] = *(const float4*)(base + 1024);
        }
    };

    auto store_tile = [&]() {
#pragma unroll
        for (int it = 0; it < 8; ++it) {
            int id  = tid + it * 256;
            int row = id >> 4;
            int c4  = (id & 15) << 2;
            PackB64 pk;
            pk.h[0] = (__bf16)(pa[it].x * pw[it]);
            pk.h[1] = (__bf16)(pa[it].y * pw[it]);
            pk.h[2] = (__bf16)(pa[it].z * pw[it]);
            pk.h[3] = (__bf16)(pa[it].w * pw[it]);
            *(unsigned long long*)&As[row * 72 + c4] = pk.u;   // ds_store_b64
        }
#pragma unroll
        for (int it = 0; it < 4; ++it) {
            int id = tid + it * 256;
            int kp = id >> 5;
            int c4 = (id & 31) << 2;
            float lo[4] = { pbl[it].x, pbl[it].y, pbl[it].z, pbl[it].w };
            float hi[4] = { pbh[it].x, pbh[it].y, pbh[it].z, pbh[it].w };
#pragma unroll
            for (int e = 0; e < 4; ++e) {     // transpose: pack (k,k+1) per o
                PackB32 pk;
                pk.h[0] = (__bf16)lo[e];
                pk.h[1] = (__bf16)hi[e];
                *(unsigned*)&Bs[(c4 + e) * 72 + 2 * kp] = pk.u; // ds_store_b32
            }
        }
    };

    prefetch(0);

#pragma unroll 1
    for (int k0 = 0; k0 < 16384; k0 += 64) {
        __syncthreads();                  // previous tile's LDS reads complete
        store_tile();
        __syncthreads();                  // LDS tile ready
        if (k0 + 64 < 16384) prefetch(k0 + 64);   // overlap with WMMAs below

#pragma unroll
        for (int ks = 0; ks < 64; ks += 32) {
            FragBF afr[4], bfr[2];
            // A fragment: lane holds M=lm, K in {8lh..8lh+7} U {16+8lh..+7}
#pragma unroll
            for (int mi = 0; mi < 4; ++mi) {
                const __bf16* s = &As[(wm + mi * 16 + lm) * 72 + ks + lh * 8];
                afr[mi].q[0] = *(const v4u*)(s);
                afr[mi].q[1] = *(const v4u*)(s + 16);
            }
            // B fragment: lane holds N=lm, K in {8lh..8lh+7} U {16+8lh..+7}
#pragma unroll
            for (int ni = 0; ni < 2; ++ni) {
                const __bf16* s = &Bs[(wn + ni * 16 + lm) * 72 + ks + lh * 8];
                bfr[ni].q[0] = *(const v4u*)(s);
                bfr[ni].q[1] = *(const v4u*)(s + 16);
            }
#pragma unroll
            for (int mi = 0; mi < 4; ++mi)
#pragma unroll
                for (int ni = 0; ni < 2; ++ni)
                    acc[mi][ni] = __builtin_amdgcn_wmma_f32_16x16x32_bf16(
                        false, afr[mi].v, false, bfr[ni].v,
                        (short)0, acc[mi][ni], false, false);
        }
    }

    // --- epilogue: C/D layout lane -> (M = r + 8*lh, N = lm); add power-weighted by ---
    float byv0[16], byv1[16];
    const int oA = o0 + wn + lm;
    const int oB = o0 + wn + 16 + lm;
#pragma unroll
    for (int n = 0; n < 16; ++n) {
        byv0[n] = by[n * 1024 + oA];
        byv1[n] = by[n * 1024 + oB];
    }
#pragma unroll
    for (int mi = 0; mi < 4; ++mi) {
#pragma unroll
        for (int r = 0; r < 8; ++r) {
            const int b = m0 + wm + mi * 16 + 8 * lh + r;
            float b0v = 0.0f, b1v = 0.0f;
            for (int n = 0; n < 16; ++n) {
                float p = power[b * 16 + n];
                b0v = fmaf(p, byv0[n], b0v);
                b1v = fmaf(p, byv1[n], b1v);
            }
            out[(size_t)b * 1024 + oA] = acc[mi][0][r] + b0v;
            out[(size_t)b * 1024 + oB] = acc[mi][1][r] + b1v;
        }
    }
}

// ---------------------------------------------------------------------------
extern "C" void kernel_launch(void* const* d_in, const int* in_sizes, int n_in,
                              void* d_out, int out_size, void* d_ws, size_t ws_size,
                              hipStream_t stream) {
    const float* x  = (const float*)d_in[0];   // [4096,1024]
    const float* Wy = (const float*)d_in[1];   // [16,1024,1024]
    const float* by = (const float*)d_in[2];   // [16,1024]
    const float* Wd = (const float*)d_in[3];   // [16,1024,16]
    const float* bd = (const float*)d_in[4];   // [16,16]
    float* out   = (float*)d_out;              // [4096,1024]
    float* power = (float*)d_ws;               // [4096,16] scratch (256 KB)

    hipLaunchKernelGGL(power_kernel, dim3(128), dim3(256), 0, stream, x, Wd, bd, power);
    hipLaunchKernelGGL(gemm_kernel, dim3(32, 8), dim3(256), 0, stream, x, Wy, by, power, out);
}
